// Encoder_82480551952478
// MI455X (gfx1250) — matrix operand order repaired and checked by reading, not compiled, not run on hardware
//
#include <hip/hip_runtime.h>

// ---------------- problem constants ----------------
#define B_      8192
#define TSTEPS  9          // T-1
#define INP_    81
#define H_      512
#define F_      405        // 5*81 conv features
#define FP_     416        // F_ padded to 16
#define FTILES  26         // FP_/16
#define KCONV   729        // 9*81 im2col K
#define KCONVP  736        // padded to 32
#define KCCH    23         // KCONVP/32
#define G4      2048       // 4*H
#define KHCH    16         // 512/32 K-chunks for Whh
#define KWI     96         // 81 padded to 96
#define KWICH   3          // 96/32

typedef __attribute__((ext_vector_type(16))) __bf16 v16bf;
typedef __attribute__((ext_vector_type(8)))  float  v8f;

__device__ __forceinline__ unsigned short f2bf_bits(float x) {
    __bf16 b = (__bf16)x;
    return __builtin_bit_cast(unsigned short, b);
}
__device__ __forceinline__ float sigm(float x) { return 1.0f / (1.0f + __expf(-x)); }

// ============ weight packing into WMMA B-fragment order ============
// B fragment (16-bit, 32x16): lane l -> N = l%16, K = (l<16?0:8)+(i<8?i:i+8), i=0..15
// stored linearly: [ntile][kchunk][lane][16 elems] -> one 32B load per lane.

__global__ void k_pack_whh(const float* __restrict__ Whh, unsigned short* __restrict__ dst) {
    int idx = blockIdx.x * 256 + threadIdx.x;
    if (idx >= G4 * H_) return;
    int e    = idx & 15;
    int lane = (idx >> 4) & 31;
    int kc   = (idx >> 9) & 15;
    int nt   = idx >> 13;
    int n = nt * 16 + (lane & 15);
    int k = kc * 32 + ((lane < 16) ? 0 : 8) + ((e < 8) ? e : e + 8);
    dst[idx] = f2bf_bits(Whh[(size_t)n * H_ + k]);
}

__global__ void k_pack_wih(const float* __restrict__ Wih, unsigned short* __restrict__ dst) {
    int idx = blockIdx.x * 256 + threadIdx.x;
    if (idx >= 128 * KWICH * 512) return;
    int e    = idx & 15;
    int lane = (idx >> 4) & 31;
    int kc   = (idx >> 9) % KWICH;
    int nt   = (idx >> 9) / KWICH;
    int n = nt * 16 + (lane & 15);
    int k = kc * 32 + ((lane < 16) ? 0 : 8) + ((e < 8) ? e : e + 8);
    float v = (k < INP_) ? Wih[(size_t)n * INP_ + k] : 0.f;
    dst[idx] = f2bf_bits(v);
}

// conv im2col weight: K = t_in*81+ic (pad 736), outcols grouped [ftile][t_out]
__global__ void k_pack_conv(const float* __restrict__ w3,  const float* __restrict__ w5,
                            const float* __restrict__ w7,  const float* __restrict__ w11,
                            const float* __restrict__ w13, unsigned short* __restrict__ dst) {
    int idx = blockIdx.x * 256 + threadIdx.x;
    if (idx >= FTILES * TSTEPS * KCCH * 512) return;
    int e    = idx & 15;
    int lane = (idx >> 4) & 31;
    int rest = idx >> 9;
    int kc   = rest % KCCH;  rest /= KCCH;
    int t    = rest % TSTEPS;
    int ft   = rest / TSTEPS;
    int f = ft * 16 + (lane & 15);
    int k = kc * 32 + ((lane < 16) ? 0 : 8) + ((e < 8) ? e : e + 8);
    float val = 0.f;
    if (f < F_ && k < KCONV) {
        int t_in = k / INP_, ic = k % INP_;
        int ki = f / INP_,  oc = f % INP_;
        int ksz = (ki == 0) ? 3 : (ki == 1) ? 5 : (ki == 2) ? 7 : (ki == 3) ? 11 : 13;
        const float* w = (ki == 0) ? w3 : (ki == 1) ? w5 : (ki == 2) ? w7 : (ki == 3) ? w11 : w13;
        int p = (ksz - 1) >> 1;
        int d = t_in - t + p;
        if (d >= 0 && d < ksz) val = w[((size_t)oc * INP_ + ic) * ksz + d];
    }
    dst[idx] = f2bf_bits(val);
}

__global__ void k_swx(const float* __restrict__ regr_W, float* __restrict__ swx) {
    int o = threadIdx.x;
    if (o < INP_) {
        float s = 0.f;
        for (int j = 0; j < F_; ++j) s += regr_W[(size_t)o * 486 + j];
        swx[o] = s;
    }
}

__global__ void k_bsum(const float* __restrict__ bih, const float* __restrict__ bhh,
                       float* __restrict__ bsum) {
    int i = blockIdx.x * 256 + threadIdx.x;
    if (i < G4) bsum[i] = bih[i] + bhh[i];
}

__global__ void k_zero_hc(float* __restrict__ h, float* __restrict__ c, int n) {
    int i = blockIdx.x * 256 + threadIdx.x;
    if (i < n) { h[i] = 0.f; c[i] = 0.f; }
}

// ============ conv GEMM: (16 x 736) x (736 x [16f x 9t]), epilogue tanh+bias+sum_t*w_t ============
__global__ __launch_bounds__(96) void k_conv_gemm(
    const float* __restrict__ inp, const unsigned short* __restrict__ wpack,
    const float* __restrict__ b3,  const float* __restrict__ b5,  const float* __restrict__ b7,
    const float* __restrict__ b11, const float* __restrict__ b13,
    const float* __restrict__ attn_conv_w, float* __restrict__ xstatic) {
    __shared__ __bf16 shA[16 * KCONVP];          // A tile, plain (m,k) bf16
    __shared__ float  shPart[3][32][8];          // per-wave partial sums over t
    int tid = threadIdx.x;
    int b0  = blockIdx.x * 16;
    int ft  = blockIdx.y;
    for (int i = tid; i < 16 * KCONVP; i += 96) {
        int m = i / KCONVP, k = i % KCONVP;
        float v = (k < KCONV) ? inp[(size_t)(b0 + m) * KCONV + k] : 0.f;
        shA[i] = (__bf16)v;
    }
    __syncthreads();
    int wave = tid >> 5, lane = tid & 31;
    int ml = lane & 15;
    int aoff = (lane < 16) ? 0 : 8;
    float psum[8];
    for (int v = 0; v < 8; ++v) psum[v] = 0.f;
    for (int tloc = 0; tloc < 3; ++tloc) {
        int t = wave * 3 + tloc;
        v8f acc;
        for (int v = 0; v < 8; ++v) acc[v] = 0.f;
        for (int kc = 0; kc < KCCH; ++kc) {
            v16bf a;
            int kb = kc * 32 + aoff;
            #pragma unroll
            for (int i = 0; i < 8; ++i) a[i] = shA[ml * KCONVP + kb + i];
            #pragma unroll
            for (int i = 0; i < 8; ++i) a[8 + i] = shA[ml * KCONVP + kb + 16 + i];
            v16bf bf = *(const v16bf*)(const void*)(wpack +
                        ((((size_t)ft * TSTEPS + t) * KCCH + kc) * 32 + lane) * 16);
            acc = __builtin_amdgcn_wmma_f32_16x16x32_bf16(false, a, false, bf, (short)0,
                                                          acc, false, false);
        }
        float wt = attn_conv_w[2 * H_ + t];
        int n = lane & 15;
        #pragma unroll
        for (int v = 0; v < 8; ++v) {
            int f = ft * 16 + n;
            float val = 0.f;
            if (f < F_) {
                int ki = f / INP_, oc = f % INP_;
                const float* bp = (ki == 0) ? b3 : (ki == 1) ? b5 : (ki == 2) ? b7
                                 : (ki == 3) ? b11 : b13;
                val = tanhf(acc[v] + bp[oc]) * wt;
            }
            psum[v] += val;
        }
    }
    for (int v = 0; v < 8; ++v) shPart[wave][lane][v] = psum[v];
    __syncthreads();
    if (wave == 0) {
        int n = lane & 15;
        int mofs = (lane < 16) ? 0 : 8;
        for (int v = 0; v < 8; ++v) {
            float s = shPart[0][lane][v] + shPart[1][lane][v] + shPart[2][lane][v];
            int m = v + mofs;
            xstatic[(size_t)(b0 + m) * FP_ + ft * 16 + n] = s;
        }
    }
}

// ============ time-invariant logits part: L0 = (x_static+acb)@Wx^T + regr_b ; y_static ============
__global__ void k_static2(const float* __restrict__ inp, const float* __restrict__ xstatic,
                          const float* __restrict__ regr_W, const float* __restrict__ regr_b,
                          const float* __restrict__ acb, const float* __restrict__ aqw,
                          const float* __restrict__ aqb,
                          float* __restrict__ L0, float* __restrict__ ystat) {
    int g = blockIdx.x * 256 + threadIdx.x;
    int b = g / 96, o = g % 96;
    if (b >= B_ || o >= INP_) return;
    float ab = acb[0];
    const float* xr = xstatic + (size_t)b * FP_;
    const float* wr = regr_W + (size_t)o * 486;
    float s = regr_b[o];
    for (int j = 0; j < F_; ++j) s += (xr[j] + ab) * wr[j];
    L0[(size_t)b * 96 + o] = s;
    float ys = aqb[0];
    const float* ir = inp + (size_t)b * (TSTEPS * INP_) + o;
    for (int t = 0; t < TSTEPS; ++t) ys += ir[t * INP_] * aqw[2 * H_ + t];
    ystat[(size_t)b * 96 + o] = ys;
}

// ============ one recurrent step: attention + softmax + WMMA LSTM gates ============
__global__ __launch_bounds__(128) void k_step(int t,
    const float* __restrict__ inp, const float* __restrict__ acw,
    const float* __restrict__ aqw, const float* __restrict__ regr_W,
    const float* __restrict__ bsum,
    const unsigned short* __restrict__ whh_pack, const unsigned short* __restrict__ wih_pack,
    const float* __restrict__ swx, const float* __restrict__ L0,
    const float* __restrict__ ystat, float* __restrict__ hbuf, float* __restrict__ cbuf,
    float* __restrict__ out_w, float* __restrict__ out_h) {
    __shared__ __bf16 sh_hb[16][H_];       // 16 KB, bf16 copy of h tile (plain m,k layout)
    __shared__ float  sh_att[16][INP_];
    __shared__ float  sh_cy[16][INP_];
    __shared__ __bf16 sh_wi[16][KWI];
    __shared__ float  red0[128], red1[128];
    __shared__ float  rs0[16], rs1[16];

    int tid = threadIdx.x;
    int b0  = blockIdx.x * 16;
    // one-shot f32->bf16 conversion of the h tile (A operand for WMMA)
    for (int i = tid; i < 16 * H_; i += 128) {
        int m = i >> 9, k = i & (H_ - 1);
        sh_hb[m][k] = (__bf16)hbuf[(size_t)(b0 + m) * H_ + k];
    }
    for (int i = tid; i < 16 * (KWI - INP_); i += 128) {
        int m = i / (KWI - INP_), j = INP_ + i % (KWI - INP_);
        sh_wi[m][j] = (__bf16)0.f;
    }
    __syncthreads();

    // s_x = h.w_h + c.w_c ; s_y = h.wy_h + c.wy_c  (16 rows x 8 partials, f32 from global)
    int row = tid >> 3, part = tid & 7;
    {
        float ax = 0.f, ay = 0.f;
        size_t hb = (size_t)(b0 + row) * H_;
        for (int k = part * 64; k < part * 64 + 64; ++k) {
            float hv = hbuf[hb + k];
            float cv = cbuf[hb + k];
            ax += hv * acw[k] + cv * acw[H_ + k];
            ay += hv * aqw[k] + cv * aqw[H_ + k];
        }
        red0[tid] = ax; red1[tid] = ay;
    }
    __syncthreads();
    if (part == 0) {
        float sx = 0.f, sy = 0.f;
        for (int p = 0; p < 8; ++p) { sx += red0[row * 8 + p]; sy += red1[row * 8 + p]; }
        rs0[row] = sx; rs1[row] = sy;
    }
    __syncthreads();
    for (int i = tid; i < 16 * INP_; i += 128) {
        int m = i / INP_, j = i % INP_;
        sh_cy[m][j] = cosf(ystat[(size_t)(b0 + m) * 96 + j] + rs1[m]);
    }
    __syncthreads();
    // logits = L0 + s_x*S_Wx + cy@Wy^T
    for (int i = tid; i < 16 * INP_; i += 128) {
        int m = i / INP_, o = i % INP_;
        float s = L0[(size_t)(b0 + m) * 96 + o] + rs0[m] * swx[o];
        const float* wy = regr_W + (size_t)o * 486 + F_;
        for (int j = 0; j < INP_; ++j) s += sh_cy[m][j] * wy[j];
        sh_att[m][o] = s;
    }
    __syncthreads();
    // softmax over 81
    {
        float mx = -1e30f;
        for (int j = part; j < INP_; j += 8) mx = fmaxf(mx, sh_att[row][j]);
        red0[tid] = mx;
    }
    __syncthreads();
    if (part == 0) {
        float mx = -1e30f;
        for (int p = 0; p < 8; ++p) mx = fmaxf(mx, red0[row * 8 + p]);
        rs0[row] = mx;
    }
    __syncthreads();
    {
        float sm = 0.f;
        for (int j = part; j < INP_; j += 8) {
            float e = __expf(sh_att[row][j] - rs0[row]);
            sh_att[row][j] = e; sm += e;
        }
        red0[tid] = sm;
    }
    __syncthreads();
    if (part == 0) {
        float sm = 0.f;
        for (int p = 0; p < 8; ++p) sm += red0[row * 8 + p];
        rs1[row] = sm;
    }
    __syncthreads();
    for (int i = tid; i < 16 * INP_; i += 128) {
        int m = i / INP_, j = i % INP_;
        float a  = sh_att[m][j] / rs1[m];
        float xt = inp[((size_t)(b0 + m) * TSTEPS + t) * INP_ + j];
        float wi = a * xt;
        out_w[((size_t)(b0 + m) * TSTEPS + t) * INP_ + j] = wi;
        sh_wi[m][j] = (__bf16)wi;
    }
    __syncthreads();

    // g = h@Whh^T + wi@Wih^T + bsum   via v_wmma_f32_16x16x32_bf16
    int wave = tid >> 5, lane = tid & 31;
    int ml = lane & 15;
    int aoff = (lane < 16) ? 0 : 8;
    for (int nt8 = 0; nt8 < 8; ++nt8) {
        int ht = wave * 8 + nt8;                       // h-column tile 0..31
        v8f acc[4];
        for (int q = 0; q < 4; ++q)
            for (int v = 0; v < 8; ++v) acc[q][v] = 0.f;
        for (int kc = 0; kc < KHCH; ++kc) {
            v16bf a;
            int kb = kc * 32 + aoff;
            #pragma unroll
            for (int i = 0; i < 8; ++i) a[i] = sh_hb[ml][kb + i];
            #pragma unroll
            for (int i = 0; i < 8; ++i) a[8 + i] = sh_hb[ml][kb + 16 + i];
            #pragma unroll
            for (int q = 0; q < 4; ++q) {
                int ntile = q * 32 + ht;
                v16bf bf = *(const v16bf*)(const void*)(whh_pack +
                            (((size_t)ntile * KHCH + kc) * 32 + lane) * 16);
                acc[q] = __builtin_amdgcn_wmma_f32_16x16x32_bf16(false, a, false, bf,
                                                                 (short)0, acc[q], false, false);
            }
        }
        for (int kc = 0; kc < KWICH; ++kc) {
            v16bf a;
            int kb = kc * 32 + aoff;
            #pragma unroll
            for (int i = 0; i < 8; ++i) a[i] = sh_wi[ml][kb + i];
            #pragma unroll
            for (int i = 0; i < 8; ++i) a[8 + i] = sh_wi[ml][kb + 16 + i];
            #pragma unroll
            for (int q = 0; q < 4; ++q) {
                int ntile = q * 32 + ht;
                v16bf bf = *(const v16bf*)(const void*)(wih_pack +
                            (((size_t)ntile * KWICH + kc) * 32 + lane) * 16);
                acc[q] = __builtin_amdgcn_wmma_f32_16x16x32_bf16(false, a, false, bf,
                                                                 (short)0, acc[q], false, false);
            }
        }
        // fused LSTM gate epilogue; D elem: m = v + (lane<16?0:8), n = lane%16
        int n = lane & 15;
        int mofs = (lane < 16) ? 0 : 8;
        int hcol = ht * 16 + n;
        #pragma unroll
        for (int v = 0; v < 8; ++v) {
            int m = v + mofs;
            size_t b = (size_t)(b0 + m);
            float gi = acc[0][v] + bsum[hcol];
            float gf = acc[1][v] + bsum[H_ + hcol];
            float gg = acc[2][v] + bsum[2 * H_ + hcol];
            float go = acc[3][v] + bsum[3 * H_ + hcol];
            float i_ = sigm(gi), f_ = sigm(gf), o_ = sigm(go);
            float g_ = tanhf(gg);
            float cold = cbuf[b * H_ + hcol];
            float c2 = f_ * cold + i_ * g_;
            float h2 = o_ * tanhf(c2);
            cbuf[b * H_ + hcol] = c2;
            hbuf[b * H_ + hcol] = h2;
            out_h[(b * TSTEPS + t) * H_ + hcol] = h2;
        }
    }
}

// ================================ host ================================
extern "C" void kernel_launch(void* const* d_in, const int* in_sizes, int n_in,
                              void* d_out, int out_size, void* d_ws, size_t ws_size,
                              hipStream_t stream) {
    (void)in_sizes; (void)n_in; (void)out_size; (void)ws_size;
    const float* inp  = (const float*)d_in[0];
    const float* w3   = (const float*)d_in[1];  const float* b3  = (const float*)d_in[2];
    const float* w5   = (const float*)d_in[3];  const float* b5  = (const float*)d_in[4];
    const float* w7   = (const float*)d_in[5];  const float* b7  = (const float*)d_in[6];
    const float* w11  = (const float*)d_in[7];  const float* b11 = (const float*)d_in[8];
    const float* w13  = (const float*)d_in[9];  const float* b13 = (const float*)d_in[10];
    const float* acw  = (const float*)d_in[11]; const float* acb = (const float*)d_in[12];
    const float* aqw  = (const float*)d_in[13]; const float* aqb = (const float*)d_in[14];
    const float* regW = (const float*)d_in[15]; const float* regb= (const float*)d_in[16];
    const float* Wih  = (const float*)d_in[17]; const float* Whh = (const float*)d_in[18];
    const float* bih  = (const float*)d_in[19]; const float* bhh = (const float*)d_in[20];

    char* ws = (char*)d_ws;
    size_t off = 0;
    auto take = [&](size_t bytes) -> char* {
        char* p = ws + off;
        off = (off + bytes + 255) & ~(size_t)255;
        return p;
    };
    unsigned short* whh_pack  = (unsigned short*)take((size_t)G4 * H_ * 2);
    unsigned short* wih_pack  = (unsigned short*)take((size_t)128 * KWICH * 512 * 2);
    unsigned short* conv_pack = (unsigned short*)take((size_t)FTILES * TSTEPS * KCCH * 512 * 2);
    float* swx     = (float*)take(128 * 4);
    float* bsum    = (float*)take((size_t)G4 * 4);
    float* xstatic = (float*)take((size_t)B_ * FP_ * 4);
    float* L0      = (float*)take((size_t)B_ * 96 * 4);
    float* ystat   = (float*)take((size_t)B_ * 96 * 4);
    float* hbuf    = (float*)take((size_t)B_ * H_ * 4);
    float* cbuf    = (float*)take((size_t)B_ * H_ * 4);

    k_pack_whh<<<(G4 * H_ + 255) / 256, 256, 0, stream>>>(Whh, whh_pack);
    k_pack_wih<<<(128 * KWICH * 512 + 255) / 256, 256, 0, stream>>>(Wih, wih_pack);
    k_pack_conv<<<(FTILES * TSTEPS * KCCH * 512 + 255) / 256, 256, 0, stream>>>(
        w3, w5, w7, w11, w13, conv_pack);
    k_swx<<<1, 128, 0, stream>>>(regW, swx);
    k_bsum<<<(G4 + 255) / 256, 256, 0, stream>>>(bih, bhh, bsum);
    k_zero_hc<<<(B_ * H_ + 255) / 256, 256, 0, stream>>>(hbuf, cbuf, B_ * H_);

    dim3 cg(B_ / 16, FTILES);
    k_conv_gemm<<<cg, 96, 0, stream>>>(inp, conv_pack, b3, b5, b7, b11, b13, acw, xstatic);
    k_static2<<<(B_ * 96 + 255) / 256, 256, 0, stream>>>(inp, xstatic, regW, regb,
                                                         acb, aqw, aqb, L0, ystat);
    float* out_w = (float*)d_out;
    float* out_h = (float*)d_out + (size_t)B_ * TSTEPS * INP_;
    for (int t = 0; t < TSTEPS; ++t) {
        k_step<<<B_ / 16, 128, 0, stream>>>(t, inp, acw, aqw, regW, bsum,
                                            whh_pack, wih_pack, swx, L0, ystat,
                                            hbuf, cbuf, out_w, out_h);
    }
}
